// ASPMSoftmax_13700945674778
// MI455X (gfx1250) — matrix operand order repaired and checked by reading, not compiled, not run on hardware
//
#include <hip/hip_runtime.h>
#include <hip/hip_bf16.h>

typedef __attribute__((ext_vector_type(16))) __bf16 v16bf;
typedef __attribute__((ext_vector_type(8)))  __bf16 v8bf;
typedef __attribute__((ext_vector_type(4)))  __bf16 v4bf;
typedef __attribute__((ext_vector_type(8)))  float  v8f;

#define B_DIM 16
#define T_DIM 2048
#define D_DIM 1024
#define ROWS  (B_DIM * T_DIM)      // 32768 GEMM rows
#define TM    128
#define TN    128
#define TKB   32                   // K per WMMA (bf16)
#define TKP   48                   // padded LDS row (bf16 units): 96B stride, 16B aligned
#define KEEP  615                  // T - int(T*0.7) = 2048 - 1433

union FragB16 { v16bf v; v8bf h[2]; };

// ---------------------------------------------------------------------------
// K0: scores[i] = b2[0]
// ---------------------------------------------------------------------------
__global__ __launch_bounds__(256) void init_scores_kernel(float* __restrict__ scores,
                                                          const float* __restrict__ b2,
                                                          int n) {
    int i = blockIdx.x * 256 + threadIdx.x;
    if (i < n) scores[i] = b2[0];
}

// ---------------------------------------------------------------------------
// K1: fused  scores += sum_e tanh( x @ W1^T + b1 )[.,e] * w2[e]
// 128x128 tile, bf16 WMMA 16x16x32, register-staged LDS double buffering.
// Wave arrangement 2M x 4N: each A/B fragment reused twice ->
// 12 ds_load_b128 per 8 WMMAs in steady state.
// ---------------------------------------------------------------------------
__global__ __launch_bounds__(256) void gemm_tanh_score_kernel(
    const float* __restrict__ x,    // [ROWS, D]
    const float* __restrict__ W1,   // [D, D] (e major, d contiguous)
    const float* __restrict__ b1,   // [D]
    const float* __restrict__ w2,   // [D]
    float* __restrict__ scores)     // [ROWS], pre-initialized with b2
{
    __shared__ __bf16 lA[2][TM][TKP];  // x tile   (rows x K), bf16, double buffered
    __shared__ __bf16 lB[2][TN][TKP];  // W1 tile  (e    x K), bf16, double buffered

    const int tid  = threadIdx.x;
    const int wave = tid >> 5;
    const int lane = tid & 31;
    const int l16  = lane & 15;
    const int half = lane >> 4;

    const int wm = wave & 3;        // M group: rows [wm*32, wm*32+32), 2 subtiles
    const int wn = wave >> 2;       // N group: cols [wn*64, wn*64+64), 4 subtiles

    const int row_base = blockIdx.x * TM;   // 256 M-blocks
    const int col_base = blockIdx.y * TN;   // 8   N-blocks

    v8f acc[8] = {};                 // acc[mi*4+j] : (M subtile mi, N subtile j)

    // per-lane column constants for the epilogue (4 N-subtiles of this wave)
    float b1v[4], w2v[4];
#pragma unroll
    for (int j = 0; j < 4; ++j) {
        int c = col_base + wn * 64 + j * 16 + l16;
        b1v[j] = b1[c];
        w2v[j] = w2[c];
    }

    // per-thread tile-load slots: 4 float4 of A + 4 float4 of B per K-step
    int rr[4], cc[4];
    const float* aptr[4];
    const float* bptr[4];
#pragma unroll
    for (int i = 0; i < 4; ++i) {
        int f = i * 256 + tid;              // 0..1023 float4 slots
        rr[i] = f >> 3;                     // row 0..127
        cc[i] = (f & 7) << 2;               // K offset 0,4,..,28
        aptr[i] = x  + (size_t)(row_base + rr[i]) * D_DIM + cc[i];
        bptr[i] = W1 + (size_t)(col_base + rr[i]) * D_DIM + cc[i];
    }

    auto load_stage = [&](int k0, float4* va, float4* vb) {
#pragma unroll
        for (int i = 0; i < 4; ++i) va[i] = *(const float4*)(aptr[i] + k0);
#pragma unroll
        for (int i = 0; i < 4; ++i) vb[i] = *(const float4*)(bptr[i] + k0);
    };
    auto store_stage = [&](int buf, const float4* va, const float4* vb) {
#pragma unroll
        for (int i = 0; i < 4; ++i) {
            v4bf pa;
            pa[0] = (__bf16)va[i].x; pa[1] = (__bf16)va[i].y;
            pa[2] = (__bf16)va[i].z; pa[3] = (__bf16)va[i].w;
            *(v4bf*)&lA[buf][rr[i]][cc[i]] = pa;
            v4bf pb;
            pb[0] = (__bf16)vb[i].x; pb[1] = (__bf16)vb[i].y;
            pb[2] = (__bf16)vb[i].z; pb[3] = (__bf16)vb[i].w;
            *(v4bf*)&lB[buf][rr[i]][cc[i]] = pb;
        }
    };

    // prologue: fill buffer 0
    float4 va[4], vb[4];
    load_stage(0, va, vb);
    store_stage(0, va, vb);
    __syncthreads();

    int cur = 0;
    for (int k0 = 0; k0 < D_DIM; k0 += TKB) {
        const bool has_next = (k0 + TKB) < D_DIM;
        // issue next tile's global loads first -- latency hidden behind WMMAs
        if (has_next) load_stage(k0 + TKB, va, vb);

        // A fragments: 16x32 bf16; lanes 0-15: K 0-7 / 16-23, lanes 16-31: K 8-15 / 24-31
        FragB16 af[2];
#pragma unroll
        for (int mi = 0; mi < 2; ++mi) {
            const __bf16* arow = &lA[cur][wm * 32 + mi * 16 + l16][0];
            af[mi].h[0] = *(const v8bf*)(arow + 8 * half);
            af[mi].h[1] = *(const v8bf*)(arow + 16 + 8 * half);
        }
        // B fragments: 32x16 bf16; lane holds column, lanes 0-15: K 0-7 / 16-23, lanes 16-31: K 8-15 / 24-31
        FragB16 bfv[4];
#pragma unroll
        for (int j = 0; j < 4; ++j) {
            const __bf16* brow = &lB[cur][wn * 64 + j * 16 + l16][0];
            bfv[j].h[0] = *(const v8bf*)(brow + 8 * half);
            bfv[j].h[1] = *(const v8bf*)(brow + 16 + 8 * half);
        }
#pragma unroll
        for (int mi = 0; mi < 2; ++mi) {
#pragma unroll
            for (int j = 0; j < 4; ++j) {
                acc[mi * 4 + j] = __builtin_amdgcn_wmma_f32_16x16x32_bf16(
                    false, af[mi].v, false, bfv[j].v, (short)0, acc[mi * 4 + j], false, false);
            }
        }

        // convert + store staged registers into the other buffer
        if (has_next) store_stage(cur ^ 1, va, vb);
        __syncthreads();
        cur ^= 1;
    }

    // ---- epilogue: tanh + dot with w2, reduce over this wave's 64 columns ----
    float partial[2][8] = {};
#pragma unroll
    for (int mi = 0; mi < 2; ++mi) {
#pragma unroll
        for (int j = 0; j < 4; ++j) {
#pragma unroll
            for (int r = 0; r < 8; ++r) {
                float z = acc[mi * 4 + j][r] + b1v[j];
                float e = __expf(2.0f * z);
                float t = 1.0f - 2.0f / (e + 1.0f);     // tanh(z)
                partial[mi][r] += t * w2v[j];
            }
        }
    }
    // reduce across the 16 lanes of each half (rows differ between halves)
#pragma unroll
    for (int m = 1; m < 16; m <<= 1) {
#pragma unroll
        for (int mi = 0; mi < 2; ++mi)
#pragma unroll
            for (int r = 0; r < 8; ++r)
                partial[mi][r] += __shfl_xor(partial[mi][r], m, 32);
    }
    if (l16 == 0) {
        // C/D layout: VGPR r -> row r (lanes 0-15) or r+8 (lanes 16-31)
#pragma unroll
        for (int mi = 0; mi < 2; ++mi) {
            int rb = row_base + wm * 32 + mi * 16 + 8 * half;
#pragma unroll
            for (int r = 0; r < 8; ++r)
                atomicAdd(&scores[rb + r], partial[mi][r]);
        }
    }
}

// ---------------------------------------------------------------------------
// K2: per-batch softmax + exact top-KEEP threshold (radix search on ordered
//     uint bits) + masked weight write. One block per batch.
// ---------------------------------------------------------------------------
__global__ __launch_bounds__(256) void softmax_mask_kernel(
    const float* __restrict__ scores, float* __restrict__ attw)
{
    __shared__ float        s[T_DIM];
    __shared__ unsigned int us[T_DIM];
    __shared__ float        red[256];
    __shared__ unsigned int cnt;

    const int b   = blockIdx.x;
    const int tid = threadIdx.x;
    const float* sb = scores + (size_t)b * T_DIM;

    float lmax = -3.402823466e38f;
    for (int t = tid; t < T_DIM; t += 256) {
        float v = sb[t];
        s[t] = v;
        unsigned u = __float_as_uint(v);
        us[t] = (u & 0x80000000u) ? ~u : (u | 0x80000000u);   // order-preserving map
        lmax = fmaxf(lmax, v);
    }
    red[tid] = lmax; __syncthreads();
    for (int o = 128; o > 0; o >>= 1) {
        if (tid < o) red[tid] = fmaxf(red[tid], red[tid + o]);
        __syncthreads();
    }
    float mx = red[0]; __syncthreads();

    float lsum = 0.0f;
    for (int t = tid; t < T_DIM; t += 256) lsum += __expf(s[t] - mx);
    red[tid] = lsum; __syncthreads();
    for (int o = 128; o > 0; o >>= 1) {
        if (tid < o) red[tid] += red[tid + o];
        __syncthreads();
    }
    float inv = 1.0f / red[0]; __syncthreads();

    // KEEP-th largest via 32-step binary search on mapped bits
    unsigned prefix = 0u;
    for (int bit = 31; bit >= 0; --bit) {
        unsigned cand = prefix | (1u << bit);
        if (tid == 0) cnt = 0u;
        __syncthreads();
        unsigned local = 0u;
        for (int t = tid; t < T_DIM; t += 256) local += (us[t] >= cand) ? 1u : 0u;
        atomicAdd(&cnt, local);
        __syncthreads();
        if (cnt >= KEEP) prefix = cand;
        __syncthreads();
    }

    for (int t = tid; t < T_DIM; t += 256) {
        float w = (us[t] >= prefix) ? __expf(s[t] - mx) * inv : 0.0f;
        attw[(size_t)b * T_DIM + t] = w;
    }
}

// ---------------------------------------------------------------------------
// K3: masked_output = x * w[bt], float4 streaming
// ---------------------------------------------------------------------------
__global__ __launch_bounds__(256) void scale_kernel(
    const float* __restrict__ x, const float* __restrict__ attw,
    float* __restrict__ out)
{
    size_t i = (size_t)blockIdx.x * 256 + threadIdx.x;   // float4 index; D/4 = 256 per row
    float  w = attw[i >> 8];
    float4 v = ((const float4*)x)[i];
    float4 o;
    o.x = v.x * w; o.y = v.y * w; o.z = v.z * w; o.w = v.w * w;
    ((float4*)out)[i] = o;
}

// ---------------------------------------------------------------------------
extern "C" void kernel_launch(void* const* d_in, const int* in_sizes, int n_in,
                              void* d_out, int out_size, void* d_ws, size_t ws_size,
                              hipStream_t stream) {
    const float* x  = (const float*)d_in[0];
    const float* W1 = (const float*)d_in[1];
    const float* b1 = (const float*)d_in[2];
    const float* w2 = (const float*)d_in[3];
    const float* b2 = (const float*)d_in[4];

    float* out    = (float*)d_out;                       // [B,T,D] masked_output
    float* attw   = out + (size_t)ROWS * D_DIM;          // [B,T]   masked weights
    float* scores = (float*)d_ws;                        // [B,T]   scratch

    init_scores_kernel<<<(ROWS + 255) / 256, 256, 0, stream>>>(scores, b2, ROWS);
    gemm_tanh_score_kernel<<<dim3(ROWS / TM, D_DIM / TN), 256, 0, stream>>>(
        x, W1, b1, w2, scores);
    softmax_mask_kernel<<<B_DIM, 256, 0, stream>>>(scores, attw);
    scale_kernel<<<(size_t)ROWS * (D_DIM / 4) / 256, 256, 0, stream>>>(x, attw, out);
}
